// ScannedRNN_88167088652620
// MI455X (gfx1250) — matrix operand order repaired and checked by reading, not compiled, not run on hardware
//
#include <hip/hip_runtime.h>
#include <hip/hip_bf16.h>

// GRU scan for MI455X (gfx1250, wave32, WMMA).
// Phase 1: xr/xz/xn = ins @ W_i{r,z,n} + b  -> bf16 workspace (memory-bound; bf16 halves traffic)
// Phase 2: sequential scan. W_hr/W_hz live in LDS (bf16, transposed), W_hn lives in VGPRs
//          (8 B-fragments), hidden state lives in LDS (bf16). Per-step reset masking in LDS.

typedef __attribute__((ext_vector_type(16))) __bf16 v16bf;
typedef __attribute__((ext_vector_type(8)))  float  v8f;

union BFrag {
    v16bf bf;
    unsigned short s[16];
    uint4 q[2];
};

static constexpr int Tt = 512;
static constexpr int Bb = 512;
static constexpr int Dd = 256;
static constexpr int Hh = 256;
static constexpr long long TB = (long long)Tt * Bb;

// LDS row strides (bf16 elements). 264*2B = 528B: 16B-aligned rows, bank-spread.
static constexpr int WSTR = 264;
static constexpr int HS   = 264;

// Native bf16 converts -> v_cvt_pk_bf16_f32 class codegen instead of integer rounding.
__device__ __forceinline__ unsigned short f2bf_bits(float f) {
    union { __bf16 h; unsigned short s; } v;
    v.h = (__bf16)f;
    return v.s;
}
__device__ __forceinline__ float bf2f(unsigned short u) {
    union { unsigned short s; __bf16 h; } v;
    v.s = u;
    return (float)v.h;
}

__device__ __forceinline__ v8f wmma_bf16(const BFrag& a, const BFrag& b, v8f c) {
    return __builtin_amdgcn_wmma_f32_16x16x32_bf16(
        /*neg_a=*/false, a.bf, /*neg_b=*/false, b.bf,
        /*c_mod=*/(short)0, c, /*reuse_a=*/false, /*reuse_b=*/false);
}

// ---------------------------------------------------------------------------
// Phase 1: input projections. Grid: TB/128 blocks, 256 threads (8 waves).
// Each wave owns a 16-row M-subtile; loops 3 gates x 16 N-tiles, K=256 in 8 chunks.
// W_g staged transposed in LDS ([n][k] bf16) so B-fragments are contiguous 32B.
// ---------------------------------------------------------------------------
__global__ __launch_bounds__(256) void gru_input_proj(
    const float* __restrict__ ins,
    const float* __restrict__ w_ir, const float* __restrict__ w_iz, const float* __restrict__ w_in,
    const float* __restrict__ b_ir, const float* __restrict__ b_iz, const float* __restrict__ b_in,
    unsigned short* __restrict__ xg)   // [3][TB][H] bf16
{
    __shared__ __align__(16) unsigned short wl[Hh * WSTR];   // 132 KB

    const int tid  = threadIdx.x;
    const int wave = tid >> 5;
    const int lane = tid & 31;
    const bool hi  = lane >= 16;

    const int Mbase = blockIdx.x * 128 + wave * 16;
    const int arow  = Mbase + (lane & 15);
    const float* ap = ins + (size_t)arow * Dd;

    // A-fragments (ins rows, full K=256): contiguous k order (lanes 0-15 K=0..15, lanes 16-31 K=16..31).
    BFrag afr[8];
#pragma unroll
    for (int c = 0; c < 8; ++c) {
        const int k0 = c * 32 + (hi ? 16 : 0);
        const int k1 = k0 + 8;
        float4 f0 = *(const float4*)(ap + k0);
        float4 f1 = *(const float4*)(ap + k0 + 4);
        float4 f2 = *(const float4*)(ap + k1);
        float4 f3 = *(const float4*)(ap + k1 + 4);
        afr[c].s[0]  = f2bf_bits(f0.x); afr[c].s[1]  = f2bf_bits(f0.y);
        afr[c].s[2]  = f2bf_bits(f0.z); afr[c].s[3]  = f2bf_bits(f0.w);
        afr[c].s[4]  = f2bf_bits(f1.x); afr[c].s[5]  = f2bf_bits(f1.y);
        afr[c].s[6]  = f2bf_bits(f1.z); afr[c].s[7]  = f2bf_bits(f1.w);
        afr[c].s[8]  = f2bf_bits(f2.x); afr[c].s[9]  = f2bf_bits(f2.y);
        afr[c].s[10] = f2bf_bits(f2.z); afr[c].s[11] = f2bf_bits(f2.w);
        afr[c].s[12] = f2bf_bits(f3.x); afr[c].s[13] = f2bf_bits(f3.y);
        afr[c].s[14] = f2bf_bits(f3.z); afr[c].s[15] = f2bf_bits(f3.w);
    }

    const float* Wg[3] = { w_ir, w_iz, w_in };
    const float* Bg[3] = { b_ir, b_iz, b_in };

#pragma unroll
    for (int g = 0; g < 3; ++g) {
        __syncthreads();   // previous gate's LDS reads complete
        // stage W_g transposed: wl[n*WSTR + k] = bf16(W[k*H + n]); coalesced global reads
        for (int idx = tid; idx < Dd * Hh; idx += 256) {
            const int k = idx >> 8;
            const int n = idx & 255;
            wl[n * WSTR + k] = f2bf_bits(Wg[g][idx]);
        }
        __syncthreads();

        unsigned short* outg = xg + (size_t)g * TB * Hh;

        for (int nt = 0; nt < 16; ++nt) {
            const int n = nt * 16 + (lane & 15);
            v8f acc = {0.f, 0.f, 0.f, 0.f, 0.f, 0.f, 0.f, 0.f};
#pragma unroll
            for (int c = 0; c < 8; ++c) {
                BFrag bfr;   // B 32x16: lanes 0-15 K=0..15, lanes 16-31 K=16..31
                const unsigned short* bp = wl + n * WSTR + c * 32 + (hi ? 16 : 0);
                bfr.q[0] = *(const uint4*)(bp);
                bfr.q[1] = *(const uint4*)(bp + 8);
                acc = wmma_bf16(afr[c], bfr, acc);
            }
            const float bias = Bg[g][n];
#pragma unroll
            for (int i = 0; i < 8; ++i) {
                const int r = Mbase + i + (hi ? 8 : 0);
                outg[(size_t)r * Hh + n] = f2bf_bits(acc[i] + bias);
            }
        }
    }
}

// ---------------------------------------------------------------------------
// Phase 2: sequential scan. Grid: B/16 blocks, 512 threads (16 waves).
// Wave w owns output columns [w*16, w*16+16). W_hr/W_hz in LDS (2 x 132 KB),
// W_hn in VGPRs (64/lane), h tile in LDS (8.25 KB). Total LDS ~272 KB < 320 KB.
// ---------------------------------------------------------------------------
__global__ __launch_bounds__(512) void gru_scan(
    const unsigned char* __restrict__ resets,  // [T][B] (bool, 1 byte)
    const float* __restrict__ h0,              // [B][H]
    const float* __restrict__ w_hr, const float* __restrict__ w_hz, const float* __restrict__ w_hn,
    const float* __restrict__ b_hn,
    const unsigned short* __restrict__ xg,     // [3][TB][H] bf16
    float* __restrict__ out)                   // [T][B][H] f32
{
    __shared__ __align__(16) unsigned short wrl[Hh * WSTR];  // gate r weights [n][k]
    __shared__ __align__(16) unsigned short wzl[Hh * WSTR];  // gate z weights [n][k]
    __shared__ __align__(16) unsigned short hl[16 * HS];     // hidden-state tile

    const int tid  = threadIdx.x;
    const int wave = tid >> 5;
    const int lane = tid & 31;
    const bool hi  = lane >= 16;
    const int m0   = blockIdx.x * 16;          // batch-row tile base
    const int ncol = wave * 16 + (lane & 15);  // owned H column

    // --- stage W_hn through LDS (coalesced), extract per-wave register fragments ---
    for (int idx = tid; idx < Dd * Hh; idx += 512) {
        const int k = idx >> 8;
        const int n = idx & 255;
        wrl[n * WSTR + k] = f2bf_bits(w_hn[idx]);   // borrow wrl as staging
    }
    __syncthreads();

    BFrag wn[8];   // gate-n B-fragments, VGPR-resident (64 VGPRs/lane)
#pragma unroll
    for (int c = 0; c < 8; ++c) {
        const unsigned short* bp = wrl + ncol * WSTR + c * 32 + (hi ? 16 : 0);
        wn[c].q[0] = *(const uint4*)(bp);
        wn[c].q[1] = *(const uint4*)(bp + 8);
    }
    __syncthreads();

    // --- stage W_hr / W_hz into LDS (persistent), init h tile from h0 ---
    for (int idx = tid; idx < Dd * Hh; idx += 512) {
        const int k = idx >> 8;
        const int n = idx & 255;
        wrl[n * WSTR + k] = f2bf_bits(w_hr[idx]);
        wzl[n * WSTR + k] = f2bf_bits(w_hz[idx]);
    }
    for (int idx = tid; idx < 16 * Hh; idx += 512) {
        const int r = idx >> 8;
        const int c = idx & 255;
        hl[r * HS + c] = f2bf_bits(h0[(size_t)(m0 + r) * Hh + c]);
    }
    const float bhn = b_hn[ncol];
    __syncthreads();

    const unsigned short* wr_p = wrl + ncol * WSTR + (hi ? 16 : 0);
    const unsigned short* wz_p = wzl + ncol * WSTR + (hi ? 16 : 0);

    for (int t = 0; t < Tt; ++t) {
        // --- reset masking: zero rows of h where episode resets ---
        {
            const unsigned char* rs = resets + (size_t)t * Bb + m0;
            for (int idx = tid; idx < 16 * Hh; idx += 512) {
                const int r = idx >> 8;
                const int c = idx & 255;
                if (rs[r]) hl[r * HS + c] = 0;
            }
        }
        __syncthreads();

        // --- prefetch next step's gate activations (global_prefetch_b8) ---
        if (t + 1 < Tt) {
            const size_t nrow = (size_t)(t + 1) * Bb + m0 + (lane & 15);
#pragma unroll
            for (int g = 0; g < 3; ++g)
                __builtin_prefetch(xg + ((size_t)g * TB + nrow) * Hh + wave * 16, 0, 3);
        }

        // --- A-fragments of masked h (shared by all 3 gates) ---
        BFrag afr[8];
        const int arow = lane & 15;
#pragma unroll
        for (int c = 0; c < 8; ++c) {
            const int k0 = c * 32 + (hi ? 16 : 0);
            const int k1 = k0 + 8;
            afr[c].q[0] = *(const uint4*)(hl + arow * HS + k0);
            afr[c].q[1] = *(const uint4*)(hl + arow * HS + k1);
        }

        // --- h @ W_h{r,z,n}: r/z B-frags stream from LDS, n from registers ---
        v8f ar = {0.f, 0.f, 0.f, 0.f, 0.f, 0.f, 0.f, 0.f};
        v8f az = {0.f, 0.f, 0.f, 0.f, 0.f, 0.f, 0.f, 0.f};
        v8f an = {0.f, 0.f, 0.f, 0.f, 0.f, 0.f, 0.f, 0.f};
#pragma unroll
        for (int c = 0; c < 8; ++c) {
            BFrag br, bz;
            br.q[0] = *(const uint4*)(wr_p + c * 32);
            br.q[1] = *(const uint4*)(wr_p + c * 32 + 8);
            bz.q[0] = *(const uint4*)(wz_p + c * 32);
            bz.q[1] = *(const uint4*)(wz_p + c * 32 + 8);
            ar = wmma_bf16(afr[c], br, ar);
            az = wmma_bf16(afr[c], bz, az);
            an = wmma_bf16(afr[c], wn[c], an);
        }

        // --- elementwise gate math + output store ---
        float hnew[8];
        const size_t trow = (size_t)t * Bb + m0;
#pragma unroll
        for (int i = 0; i < 8; ++i) {
            const int r = i + (hi ? 8 : 0);
            const size_t grow = trow + r;
            const float xr = bf2f(xg[((size_t)0 * TB + grow) * Hh + ncol]);
            const float xz = bf2f(xg[((size_t)1 * TB + grow) * Hh + ncol]);
            const float xn = bf2f(xg[((size_t)2 * TB + grow) * Hh + ncol]);
            const float hp = bf2f(hl[r * HS + ncol]);   // masked h_prev
            const float rg = 1.f / (1.f + __expf(-(xr + ar[i])));
            const float zg = 1.f / (1.f + __expf(-(xz + az[i])));
            const float ng = tanhf(xn + rg * (an[i] + bhn));
            const float hv = (1.f - zg) * ng + zg * hp;
            hnew[i] = hv;
            out[grow * Hh + ncol] = hv;
        }
        __syncthreads();   // all hl reads (A-frags + hp) complete

#pragma unroll
        for (int i = 0; i < 8; ++i) {
            const int r = i + (hi ? 8 : 0);
            hl[r * HS + ncol] = f2bf_bits(hnew[i]);
        }
        __syncthreads();   // h_new visible before next step's masking
    }
}

extern "C" void kernel_launch(void* const* d_in, const int* in_sizes, int n_in,
                              void* d_out, int out_size, void* d_ws, size_t ws_size,
                              hipStream_t stream) {
    const float*         ins    = (const float*)d_in[0];
    const unsigned char* resets = (const unsigned char*)d_in[1];  // numpy bool_: 1 byte
    const float*         h0     = (const float*)d_in[2];
    const float*         w_ir   = (const float*)d_in[3];
    const float*         w_iz   = (const float*)d_in[4];
    const float*         w_in   = (const float*)d_in[5];
    const float*         b_ir   = (const float*)d_in[6];
    const float*         b_iz   = (const float*)d_in[7];
    const float*         b_in   = (const float*)d_in[8];
    const float*         w_hr   = (const float*)d_in[9];
    const float*         w_hz   = (const float*)d_in[10];
    const float*         w_hn   = (const float*)d_in[11];
    const float*         b_hn   = (const float*)d_in[12];

    unsigned short* xg  = (unsigned short*)d_ws;   // 3 * TB * H bf16 = ~403 MB scratch
    float*          out = (float*)d_out;

    gru_input_proj<<<(int)(TB / 128), 256, 0, stream>>>(
        ins, w_ir, w_iz, w_in, b_ir, b_iz, b_in, xg);

    gru_scan<<<Bb / 16, 512, 0, stream>>>(
        resets, h0, w_hr, w_hz, w_hn, b_hn, xg, out);
}
